// CrossAttentionMultiHead_1039382086261
// MI455X (gfx1250) — matrix operand order repaired and checked by reading, not compiled, not run on hardware
//
#include <hip/hip_runtime.h>
#include <hip/hip_bf16.h>

// ---------------------------------------------------------------------------
// CrossAttentionMultiHead on MI455X (gfx1250, wave32, WMMA bf16)
// B=32 C=512 H=W=32 (N=1024), TXT=768, M=77 (pad 96), heads=8, d=64
// ---------------------------------------------------------------------------

typedef __attribute__((ext_vector_type(16))) __bf16 v16bf;
typedef __attribute__((ext_vector_type(8)))  __bf16 v8bf;
typedef __attribute__((ext_vector_type(8)))  float  v8f;

#define BB 32
#define CC 512
#define NN 1024
#define TXT 768
#define MM 77
#define MP 96          // padded M (3 k-steps of 32 for P@V)
#define NH 8
#define HD 64
#define ATT_SCALE 0.125f

// ---- WMMA helpers ----------------------------------------------------------
__device__ __forceinline__ v8f wmma_bf16(v16bf a, v16bf b, v8f c) {
  // (neg_a, A, neg_b, B, c_mod, C, reuse_a, reuse_b)
  return __builtin_amdgcn_wmma_f32_16x16x32_bf16(false, a, false, b, (short)0, c,
                                                 false, false);
}

// A-fragment 16x32 bf16 (row-major MxK). rowptr -> element (row, kk).
// lane<16 : row=lane,    K = {kb..kb+15}, kb=0
// lane>=16: row=lane-16, kb=16
__device__ __forceinline__ v16bf frag_a(const __bf16* rowptr, int half) {
  union { v16bf v; v8bf h[2]; } u;
  const __bf16* p = rowptr + half * 16;
  u.h[0] = *(const v8bf*)(p);
  u.h[1] = *(const v8bf*)(p + 8);
  return u.v;
}

// B-fragment 32x16 bf16 stored [N][K] (K-contiguous). colptr -> (col, kk).
// lane holds column l&15, 16 contiguous K starting at (l>>4)*16.
__device__ __forceinline__ v16bf frag_b(const __bf16* colptr, int half) {
  return *(const v16bf*)(colptr + half * 16);
}

// ---- Pack kernels ----------------------------------------------------------
// x [B][C][N] f32 -> xT [B][N][C] bf16 (LDS tile transpose, conflict-free)
__global__ void __launch_bounds__(256) xpose_kernel(const float* __restrict__ x,
                                                    __bf16* __restrict__ xT) {
  __shared__ float tile[32][33];
  const int b  = blockIdx.z;
  const int n0 = blockIdx.x * 32;   // N
  const int c0 = blockIdx.y * 32;   // C
  const int tx = threadIdx.x, ty = threadIdx.y;   // 32 x 8
  const float* xb = x + (size_t)b * CC * NN;
#pragma unroll
  for (int k = 0; k < 4; ++k)
    tile[ty + k * 8][tx] = xb[(size_t)(c0 + ty + k * 8) * NN + n0 + tx];
  __syncthreads();
  __bf16* xTb = xT + (size_t)b * NN * CC;
#pragma unroll
  for (int k = 0; k < 4; ++k)
    xTb[(size_t)(n0 + ty + k * 8) * CC + c0 + tx] = (__bf16)tile[tx][ty + k * 8];
}

// text [B][77][768] f32 -> tT [B][96][768] bf16, rows 77..95 zeroed
__global__ void __launch_bounds__(256) textpack_kernel(const float* __restrict__ t,
                                                       __bf16* __restrict__ tT) {
  int idx = blockIdx.x * 256 + threadIdx.x;
  if (idx >= BB * MP * TXT) return;
  int k = idx % TXT, rest = idx / TXT;
  int m = rest % MP, b = rest / MP;
  float v = (m < MM) ? t[((size_t)b * MM + m) * TXT + k] : 0.0f;
  tT[idx] = (__bf16)v;
}

__global__ void __launch_bounds__(256) cvt_kernel(const float* __restrict__ s,
                                                  __bf16* __restrict__ d, int n) {
  int i = blockIdx.x * 256 + threadIdx.x;
  if (i < n) d[i] = (__bf16)s[i];
}

// ---- GEMM: Q = xT @ Wq^T + bq  -> Qb bf16 [B][N][C] ------------------------
// One wave computes a 16x64 tile; K=512 in 16 steps of 32; 4 WMMA per step.
__global__ void __launch_bounds__(128) qproj_kernel(const __bf16* __restrict__ xT,
                                                    const __bf16* __restrict__ Wqb,
                                                    const float* __restrict__ bq,
                                                    __bf16* __restrict__ Qb) {
  const int lane = threadIdx.x & 31, wave = threadIdx.x >> 5;
  int tile = blockIdx.x * 4 + wave;          // 32 * 64 * 8 tiles
  const int ot = tile & 7;  tile >>= 3;
  const int nt = tile & 63; tile >>= 6;
  const int b  = tile;
  const int l15 = lane & 15, half = lane >> 4;
  const int o0 = ot * 64;
  const __bf16* arow = xT + ((size_t)(b * NN + nt * 16 + l15)) * CC;
  v8f acc[4] = {};
  for (int kk = 0; kk < CC; kk += 32) {
    v16bf a = frag_a(arow + kk, half);
#pragma unroll
    for (int j = 0; j < 4; ++j) {
      const int col = o0 + j * 16 + l15;
      acc[j] = wmma_bf16(a, frag_b(Wqb + (size_t)col * CC + kk, half), acc[j]);
    }
  }
#pragma unroll
  for (int j = 0; j < 4; ++j) {
    const int col = o0 + j * 16 + l15;
    const float bias = bq[col];
#pragma unroll
    for (int r = 0; r < 8; ++r) {
      const int n = nt * 16 + half * 8 + r;
      Qb[((size_t)(b * NN + n)) * CC + col] = (__bf16)(acc[j][r] + bias);
    }
  }
}

// ---- GEMM: K = tT @ Wk^T + bk -> Kb [B][96][512] (m-major)
//            V = tT @ Wv^T + bv -> Vt [B][512][96] (d-major, m-contig) -------
__global__ void __launch_bounds__(128) kvproj_kernel(const __bf16* __restrict__ tT,
                                                     const __bf16* __restrict__ Wkb,
                                                     const __bf16* __restrict__ Wvb,
                                                     const float* __restrict__ bk,
                                                     const float* __restrict__ bv,
                                                     __bf16* __restrict__ Kb,
                                                     __bf16* __restrict__ Vt) {
  const int lane = threadIdx.x & 31, wave = threadIdx.x >> 5;
  int tile = blockIdx.x * 4 + wave;          // 32 * 6 * 8 * 2 tiles
  const int kv = tile & 1; tile >>= 1;
  const int ot = tile & 7; tile >>= 3;
  const int mt = tile % 6;
  const int b  = tile / 6;
  const int l15 = lane & 15, half = lane >> 4;
  const __bf16* arow = tT + ((size_t)(b * MP + mt * 16 + l15)) * TXT;
  const __bf16* W = kv ? Wvb : Wkb;
  const float*  bias = kv ? bv : bk;
  const int o0 = ot * 64;
  v8f acc[4] = {};
  for (int kk = 0; kk < TXT; kk += 32) {
    v16bf a = frag_a(arow + kk, half);
#pragma unroll
    for (int j = 0; j < 4; ++j) {
      const int col = o0 + j * 16 + l15;
      acc[j] = wmma_bf16(a, frag_b(W + (size_t)col * TXT + kk, half), acc[j]);
    }
  }
#pragma unroll
  for (int j = 0; j < 4; ++j) {
    const int col = o0 + j * 16 + l15;
    const float bi = bias[col];
#pragma unroll
    for (int r = 0; r < 8; ++r) {
      const int m = mt * 16 + half * 8 + r;
      const float v = acc[j][r] + bi;
      if (kv) Vt[((size_t)(b * CC + col)) * MP + m] = (__bf16)v;
      else    Kb[((size_t)(b * MP + m)) * CC + col] = (__bf16)v;
    }
  }
}

// ---- Fused attention: S = Q K^T * scale, softmax(mask), out = P V ----------
// One wave per (b, head, 16-row n-tile). 24 WMMA per wave.
__global__ void __launch_bounds__(128) attn_kernel(const __bf16* __restrict__ Qb,
                                                   const __bf16* __restrict__ Kb,
                                                   const __bf16* __restrict__ Vt,
                                                   const int* __restrict__ amask,
                                                   float* __restrict__ out) {
  __shared__ __attribute__((aligned(32))) float  Sl[4][16][MP];
  __shared__ __attribute__((aligned(32))) __bf16 Pl[4][16][MP];
  const int lane = threadIdx.x & 31, w = threadIdx.x >> 5;
  int tile = blockIdx.x * 4 + w;             // 32 * 8 * 64 tiles
  const int nt = tile & 63; tile >>= 6;
  const int h  = tile & 7;  tile >>= 3;
  const int b  = tile;
  const int l15 = lane & 15, half = lane >> 4;
  const int n0 = nt * 16;
  const float NEG_INF = -__builtin_inff();

  // Q A-fragments for the full head dim (2 k-steps of 32)
  const __bf16* qrow = Qb + ((size_t)(b * NN + n0 + l15)) * CC + h * HD;
  const v16bf aq0 = frag_a(qrow, half);
  const v16bf aq1 = frag_a(qrow + 32, half);

  // scores: 6 column tiles of 16 over padded M=96
  for (int mt = 0; mt < 6; ++mt) {
    const int mcol = mt * 16 + l15;
    const __bf16* kcol = Kb + ((size_t)(b * MP + mcol)) * CC + h * HD;
    v8f acc = {};
    acc = wmma_bf16(aq0, frag_b(kcol, half), acc);
    acc = wmma_bf16(aq1, frag_b(kcol + 32, half), acc);
#pragma unroll
    for (int r = 0; r < 8; ++r) Sl[w][half * 8 + r][mcol] = acc[r];
  }
  __syncthreads();

  // masked, scaled softmax across M: lane handles row l15, columns half*48..+47
  {
    const int row = l15, c0 = half * 48;
    float mx = NEG_INF;
    for (int j = 0; j < 48; ++j) {
      const int m = c0 + j;
      const bool valid = (m < MM) && (amask[b * MM + m] != 0);
      const float s = valid ? Sl[w][row][m] * ATT_SCALE : NEG_INF;
      Sl[w][row][m] = s;
      mx = fmaxf(mx, s);
    }
    mx = fmaxf(mx, __shfl_xor(mx, 16, 32));
    float sum = 0.0f;
    for (int j = 0; j < 48; ++j) {
      const int m = c0 + j;
      const float s = Sl[w][row][m];
      const float e = (s == NEG_INF) ? 0.0f : __expf(s - mx);
      Sl[w][row][m] = e;
      sum += e;
    }
    sum += __shfl_xor(sum, 16, 32);
    const float inv = 1.0f / sum;
    for (int j = 0; j < 48; ++j) {
      const int m = c0 + j;
      Pl[w][row][m] = (__bf16)(Sl[w][row][m] * inv);
    }
  }
  __syncthreads();

  // out = P @ V : A = P [16 x 96] (3 k-steps), B = Vt columns (4 d-tiles)
  v16bf ap[3];
  const __bf16* prow = &Pl[w][l15][0];
#pragma unroll
  for (int s = 0; s < 3; ++s) ap[s] = frag_a(prow + s * 32, half);
#pragma unroll
  for (int j = 0; j < 4; ++j) {
    const int o = h * HD + j * 16 + l15;            // channel index
    const __bf16* vcol = Vt + ((size_t)(b * CC + o)) * MP;
    v8f acc = {};
#pragma unroll
    for (int s = 0; s < 3; ++s)
      acc = wmma_bf16(ap[s], frag_b(vcol + s * 32, half), acc);
#pragma unroll
    for (int r = 0; r < 8; ++r) {
      const int n = n0 + half * 8 + r;
      out[((size_t)(b * CC + o)) * NN + n] = acc[r];  // [B][C][H*W] fp32
    }
  }
}

// ---------------------------------------------------------------------------
extern "C" void kernel_launch(void* const* d_in, const int* in_sizes, int n_in,
                              void* d_out, int out_size, void* d_ws, size_t ws_size,
                              hipStream_t stream) {
  const float* x    = (const float*)d_in[0];
  const float* text = (const float*)d_in[1];
  const int*   amsk = (const int*)d_in[2];
  const float* Wq   = (const float*)d_in[3];
  const float* bq   = (const float*)d_in[4];
  const float* Wk   = (const float*)d_in[5];
  const float* bk   = (const float*)d_in[6];
  const float* Wv   = (const float*)d_in[7];
  const float* bv   = (const float*)d_in[8];
  float* out = (float*)d_out;

  // workspace carve-out (all 32B-aligned offsets), ~80.2 MB total
  char* ws = (char*)d_ws;
  __bf16* xT  = (__bf16*)(ws + 0);                        // 32 MB  [B][N][C]
  __bf16* Qb  = (__bf16*)(ws + 33554432);                 // 32 MB  [B][N][C]
  __bf16* tT  = (__bf16*)(ws + 67108864);                 // 4.5 MB [B][96][768]
  __bf16* Kb  = (__bf16*)(ws + 71827456);                 // 3 MB   [B][96][512]
  __bf16* Vt  = (__bf16*)(ws + 74973184);                 // 3 MB   [B][512][96]
  __bf16* Wqb = (__bf16*)(ws + 78118912);                 // 512 KB
  __bf16* Wkb = (__bf16*)(ws + 78643200);                 // 768 KB
  __bf16* Wvb = (__bf16*)(ws + 79429632);                 // 768 KB

  // 1) pack inputs to bf16 (transpose x, pad text, convert weights)
  xpose_kernel<<<dim3(NN / 32, CC / 32, BB), dim3(32, 8), 0, stream>>>(x, xT);
  textpack_kernel<<<(BB * MP * TXT + 255) / 256, 256, 0, stream>>>(text, tT);
  cvt_kernel<<<(CC * CC + 255) / 256, 256, 0, stream>>>(Wq, Wqb, CC * CC);
  cvt_kernel<<<(CC * TXT + 255) / 256, 256, 0, stream>>>(Wk, Wkb, CC * TXT);
  cvt_kernel<<<(CC * TXT + 255) / 256, 256, 0, stream>>>(Wv, Wvb, CC * TXT);

  // 2) projections (WMMA bf16, f32 accumulate, bias fused)
  qproj_kernel<<<(BB * (NN / 16) * (CC / 64)) / 4, 128, 0, stream>>>(xT, Wqb, bq, Qb);
  kvproj_kernel<<<(BB * (MP / 16) * (CC / 64) * 2) / 4, 128, 0, stream>>>(
      tT, Wkb, Wvb, bk, bv, Kb, Vt);

  // 3) fused attention (scores -> softmax -> P@V, WMMA both GEMMs)
  attn_kernel<<<(BB * NH * (NN / 16)) / 4, 128, 0, stream>>>(Qb, Kb, Vt, amsk, out);
}
